// GRELU_53197464928903
// MI455X (gfx1250) — matrix operand order repaired and checked by reading, not compiled, not run on hardware
//
#include <hip/hip_runtime.h>
#include <hip/hip_bf16.h>
#include <math.h>

typedef _Float16 half16 __attribute__((ext_vector_type(16)));
typedef float    float8 __attribute__((ext_vector_type(8)));

#define EMB 64
#define NUM_GRAPHS 512
#define NODE_TILE 16

// ---------- monotonic float<->uint encoding for atomicMax on floats ----------
__device__ __forceinline__ unsigned enc_f(float f) {
    unsigned b = __float_as_uint(f);
    return (b & 0x80000000u) ? ~b : (b | 0x80000000u);
}
__device__ __forceinline__ float dec_f(unsigned u) {
    unsigned b = (u & 0x80000000u) ? (u & 0x7FFFFFFFu) : ~u;
    return __uint_as_float(b);
}

// ---------- workspace init: deg=1 (self loop), pooled/cnt/scalars = 0 ----------
__global__ void init_kernel(float* deg, float* pooled, float* cnt, unsigned* scal,
                            int nNodes) {
    long long total = nNodes > (NUM_GRAPHS * EMB) ? nNodes : (NUM_GRAPHS * EMB);
    for (long long i = (long long)blockIdx.x * blockDim.x + threadIdx.x;
         i < total; i += (long long)gridDim.x * blockDim.x) {
        if (i < nNodes)            deg[i]    = 1.0f;       // self-loop degree
        if (i < NUM_GRAPHS * EMB)  pooled[i] = 0.0f;
        if (i < NUM_GRAPHS)        cnt[i]    = 0.0f;
        if (i < 64)                scal[i]   = 0u;         // gmax key=0, gsum=0.0f
    }
}

// ---------- degree accumulation over edges ----------
__global__ void deg_kernel(const int* __restrict__ dst, float* __restrict__ deg,
                           long long nEdges) {
    for (long long e = (long long)blockIdx.x * blockDim.x + threadIdx.x;
         e < nEdges; e += (long long)gridDim.x * blockDim.x) {
        atomicAdd(&deg[dst[e]], 1.0f);
    }
}

// ---------- deg -> dinv (in place); deg >= 1 always (self loop) ----------
__global__ void dinv_kernel(float* deg, int nNodes) {
    for (long long i = (long long)blockIdx.x * blockDim.x + threadIdx.x;
         i < nNodes; i += (long long)gridDim.x * blockDim.x) {
        deg[i] = rsqrtf(deg[i]);
    }
}

// ---------- agg init with self-loop term: agg = x * dinv^2 ----------
__global__ void agg_init_kernel(const float* __restrict__ x,
                                const float* __restrict__ dinv,
                                float* __restrict__ agg, long long total) {
    for (long long idx = (long long)blockIdx.x * blockDim.x + threadIdx.x;
         idx < total; idx += (long long)gridDim.x * blockDim.x) {
        long long i = idx >> 6;
        float dv = dinv[i];
        agg[idx] = x[idx] * dv * dv;
    }
}

// ---------- edge scatter: one wave per edge, float2 per lane ----------
__global__ void scatter_kernel(const int* __restrict__ src, const int* __restrict__ dst,
                               const float* __restrict__ x, const float* __restrict__ dinv,
                               float* __restrict__ agg, long long nEdges) {
    const float2* x2 = (const float2*)x;
    long long total = nEdges << 5;  // 32 lanes per edge
    for (long long gid = (long long)blockIdx.x * blockDim.x + threadIdx.x;
         gid < total; gid += (long long)gridDim.x * blockDim.x) {
        long long e = gid >> 5;
        int lane = (int)(gid & 31);
        int s = src[e], d = dst[e];
        float nrm = dinv[s] * dinv[d];
        float2 v = x2[(long long)s * 32 + lane];
        float* ap = agg + (long long)d * EMB + lane * 2;
        atomicAdd(ap,     v.x * nrm);
        atomicAdd(ap + 1, v.y * nrm);
    }
}

// ---------- diffuse (in place over agg) + graph pooling atomics ----------
__global__ void diffuse_pool_kernel(const float* __restrict__ x,
                                    const int* __restrict__ batch,
                                    float* __restrict__ agg,   // in: agg, out: diffuse_x
                                    float* __restrict__ pooled, float* __restrict__ cnt,
                                    long long total) {
    for (long long idx = (long long)blockIdx.x * blockDim.x + threadIdx.x;
         idx < total; idx += (long long)gridDim.x * blockDim.x) {
        long long i = idx >> 6;
        int c = (int)(idx & 63);
        float d = 0.9f * agg[idx] + 0.1f * x[idx];
        agg[idx] = d;
        int b = batch[i];
        atomicAdd(&pooled[(long long)b * EMB + c], d);
        if (c == 0) atomicAdd(&cnt[b], 1.0f);
    }
}

// ---------- node MLP via WMMA: h = relu(diffuse@wn1+bn1); graw = h@wn2+bn2 ----------
// Block: 128 threads (4 waves), 16 nodes per block. Wave w computes output
// columns [16w, 16w+16). K=64 handled as two 16x16x32 f16 WMMAs.
__global__ void node_mlp_kernel(const float* __restrict__ diffuse,
                                const float* __restrict__ wn1, const float* __restrict__ bn1,
                                const float* __restrict__ wn2, const float* __restrict__ bn2,
                                float* __restrict__ graw, int nNodes) {
    const int wave = threadIdx.x >> 5;
    const int lane = threadIdx.x & 31;
    const int nodeBase = blockIdx.x * NODE_TILE;
    const int row = lane & 15;                 // A-matrix M per ISA layout
    const int akb = (lane < 16) ? 0 : 8;       // unused; A loads use bkb (k = 16h + e)
    const int col = (wave << 4) + (lane & 15); // output column (N) for this lane
    const int bkb = (lane < 16) ? 0 : 16;      // K-base for A and B (contiguous k = 16h + e)

    int nodeRow = nodeBase + row;
    if (nodeRow > nNodes - 1) nodeRow = nNodes - 1;
    const float* drow = diffuse + (long long)nodeRow * EMB;

    half16 a_lo, a_hi;
#pragma unroll
    for (int e = 0; e < 16; ++e) {
        int v = e >> 1;
        int k = bkb + e;
        a_lo[e] = (_Float16)drow[k];
        a_hi[e] = (_Float16)drow[k + 32];
    }
    half16 b_lo, b_hi;
#pragma unroll
    for (int e = 0; e < 16; ++e) {
        int k = bkb + e;
        b_lo[e] = (_Float16)wn1[(long long)k * EMB + col];
        b_hi[e] = (_Float16)wn1[(long long)(k + 32) * EMB + col];
    }

    float8 c = {};
    c = __builtin_amdgcn_wmma_f32_16x16x32_f16(false, a_lo, false, b_lo, (short)0, c, false, false);
    c = __builtin_amdgcn_wmma_f32_16x16x32_f16(false, a_hi, false, b_hi, (short)0, c, false, false);

    // bias + relu + reduce h@wn2 across columns
    float bias = bn1[col];
    float w2   = wn2[col];

    __shared__ float gpart[NODE_TILE];
    if (threadIdx.x < NODE_TILE) gpart[threadIdx.x] = 0.0f;
    __syncthreads();

#pragma unroll
    for (int r = 0; r < 8; ++r) {
        float h = fmaxf(c[r] + bias, 0.0f);
        float p = h * w2;
#pragma unroll
        for (int off = 1; off < 16; off <<= 1)
            p += __shfl_xor(p, off, 32);       // reduce within each 16-lane half
        if ((lane & 15) == 0) {
            int m = r + ((lane < 16) ? 0 : 8); // C/D layout: VGPR r -> M=r / r+8
            atomicAdd(&gpart[m], p);
        }
    }
    __syncthreads();
    if (threadIdx.x < NODE_TILE && nodeBase + (int)threadIdx.x < nNodes)
        graw[nodeBase + threadIdx.x] = gpart[threadIdx.x] + bn2[0];
}

// ---------- global softmax reductions over graw ----------
__global__ void gmax_kernel(const float* __restrict__ graw, unsigned* __restrict__ scal,
                            int nNodes) {
    __shared__ float red[256];
    float m = -3.402823466e38f;
    for (long long i = (long long)blockIdx.x * blockDim.x + threadIdx.x;
         i < nNodes; i += (long long)gridDim.x * blockDim.x)
        m = fmaxf(m, graw[i]);
    red[threadIdx.x] = m;
    __syncthreads();
    for (int s = 128; s > 0; s >>= 1) {
        if (threadIdx.x < s) red[threadIdx.x] = fmaxf(red[threadIdx.x], red[threadIdx.x + s]);
        __syncthreads();
    }
    if (threadIdx.x == 0) atomicMax(&scal[0], enc_f(red[0]));
}

__global__ void gsum_kernel(const float* __restrict__ graw, unsigned* __restrict__ scal,
                            int nNodes) {
    __shared__ float red[256];
    const float gmax = dec_f(scal[0]);
    float s = 0.0f;
    for (long long i = (long long)blockIdx.x * blockDim.x + threadIdx.x;
         i < nNodes; i += (long long)gridDim.x * blockDim.x)
        s += expf(graw[i] - gmax);
    red[threadIdx.x] = s;
    __syncthreads();
    for (int t = 128; t > 0; t >>= 1) {
        if (threadIdx.x < t) red[threadIdx.x] += red[threadIdx.x + t];
        __syncthreads();
    }
    if (threadIdx.x == 0) atomicAdd((float*)&scal[1], red[0]);
}

// ---------- classifier MLP per graph: pc = tanh(relu(mean@wc1+bc1)@wc2+bc2)[:, :128] ----------
__global__ void cls_kernel(const float* __restrict__ pooled, const float* __restrict__ cnt,
                           const float* __restrict__ wc1, const float* __restrict__ bc1,
                           const float* __restrict__ wc2, const float* __restrict__ bc2,
                           float* __restrict__ pc) {
    __shared__ float sm[EMB];
    __shared__ float hid[EMB];
    const int g = blockIdx.x;
    const int t = threadIdx.x;           // 0..127
    const float cv = cnt[g];
    if (t < EMB) {
        float s = pooled[(long long)g * EMB + t];
        sm[t] = (cv > 0.0f) ? s / fmaxf(cv, 1.0f) : 0.0f;
    }
    __syncthreads();
    if (t < EMB) {
        float acc = bc1[t];
        for (int k = 0; k < EMB; ++k) acc += sm[k] * wc1[k * EMB + t];
        hid[t] = fmaxf(acc, 0.0f);
    }
    __syncthreads();
    {   // only first K*C = 128 of 256 output columns are used (alpha == beta)
        float acc = bc2[t];
        for (int k = 0; k < EMB; ++k) acc += hid[k] * wc2[k * 256 + t];
        pc[(long long)g * 128 + t] = tanhf(acc);
    }
}

// ---------- final: out = max_k( alpha_k[batch][c] * gamma_i * (x+1) ) ----------
__global__ void out_kernel(const float* __restrict__ x, const int* __restrict__ batch,
                           const float* __restrict__ graw, const float* __restrict__ pc,
                           const unsigned* __restrict__ scal, float* __restrict__ out,
                           long long total) {
    const float gmax = dec_f(scal[0]);
    const float ginv = 1.0f / ((const float*)scal)[1];
    for (long long idx = (long long)blockIdx.x * blockDim.x + threadIdx.x;
         idx < total; idx += (long long)gridDim.x * blockDim.x) {
        long long i = idx >> 6;
        int c = (int)(idx & 63);
        float g = expf(graw[i] - gmax) * ginv;
        float tt = g * (x[idx] + 1.0f);
        int b = batch[i];
        float a0 = pc[(long long)b * 128 + c];
        float a1 = pc[(long long)b * 128 + 64 + c];
        out[idx] = fmaxf(a0 * tt, a1 * tt);
    }
}

extern "C" void kernel_launch(void* const* d_in, const int* in_sizes, int n_in,
                              void* d_out, int out_size, void* d_ws, size_t ws_size,
                              hipStream_t stream) {
    const float* x    = (const float*)d_in[0];
    const int*   ei   = (const int*)  d_in[1];   // (2, E) row-major: src then dst
    const int*   batch= (const int*)  d_in[3];
    const float* wn1  = (const float*)d_in[4];
    const float* bn1  = (const float*)d_in[5];
    const float* wn2  = (const float*)d_in[6];
    const float* bn2  = (const float*)d_in[7];
    const float* wc1  = (const float*)d_in[8];
    const float* bc1  = (const float*)d_in[9];
    const float* wc2  = (const float*)d_in[10];
    const float* bc2  = (const float*)d_in[11];

    const int nNodes = in_sizes[0] / EMB;
    const long long nEdges = in_sizes[1] / 2;
    const int* src = ei;
    const int* dst = ei + nEdges;

    // workspace carve-up (floats)
    float* ws      = (float*)d_ws;
    float* deg     = ws;                         // nNodes (becomes dinv)
    float* agg     = ws + 102400;                // nNodes*64 (becomes diffuse_x)
    float* graw    = agg + (long long)102400 * EMB; // nNodes  (6,553,600 floats reserved)
    float* pooled  = graw + 102400;              // 512*64
    float* cnt     = pooled + NUM_GRAPHS * EMB;  // 512
    float* pc      = cnt + NUM_GRAPHS;           // 512*128
    unsigned* scal = (unsigned*)(pc + NUM_GRAPHS * 128); // 64 scalars

    const long long nTot = (long long)nNodes * EMB;

    init_kernel<<<256, 256, 0, stream>>>(deg, pooled, cnt, scal, nNodes);

    {
        long long b = (nEdges + 255) / 256; if (b > 65536) b = 65536;
        deg_kernel<<<(unsigned)b, 256, 0, stream>>>(dst, deg, nEdges);
    }
    dinv_kernel<<<(nNodes + 255) / 256, 256, 0, stream>>>(deg, nNodes);
    {
        long long b = (nTot + 255) / 256;
        agg_init_kernel<<<(unsigned)b, 256, 0, stream>>>(x, deg, agg, nTot);
    }
    {
        long long total = nEdges << 5;
        long long b = (total + 255) / 256; if (b > 400000) b = 400000;
        scatter_kernel<<<(unsigned)b, 256, 0, stream>>>(src, dst, x, deg, agg, nEdges);
    }
    {
        long long b = (nTot + 255) / 256;
        diffuse_pool_kernel<<<(unsigned)b, 256, 0, stream>>>(x, batch, agg, pooled, cnt, nTot);
    }
    node_mlp_kernel<<<(nNodes + NODE_TILE - 1) / NODE_TILE, 128, 0, stream>>>(
        agg, wn1, bn1, wn2, bn2, graw, nNodes);

    gmax_kernel<<<256, 256, 0, stream>>>(graw, scal, nNodes);
    gsum_kernel<<<256, 256, 0, stream>>>(graw, scal, nNodes);

    cls_kernel<<<NUM_GRAPHS, 128, 0, stream>>>(pooled, cnt, wc1, bc1, wc2, bc2, pc);

    {
        long long b = (nTot + 255) / 256;
        out_kernel<<<(unsigned)b, 256, 0, stream>>>(x, batch, graw, pc, scal,
                                                    (float*)d_out, nTot);
    }
}
